// VanillaRNN_63453846831596
// MI455X (gfx1250) — compile-verified, matrix operand order checked by reading, not run on hardware
//
#include <hip/hip_runtime.h>

// Problem sizes (fixed by the reference)
#define BB   256   // BATCH
#define SS   512   // SEQ_LEN
#define DD   512   // DIM
#define HH   1024  // HIDDEN

#define NBLK 32    // persistent scan: 8 x 4 blocks, must all be co-resident

typedef __attribute__((ext_vector_type(16))) __bf16 v16bf;
typedef __attribute__((ext_vector_type(8)))  __bf16 v8bf;
typedef __attribute__((ext_vector_type(8)))  float  v8f;

// ---------------------------------------------------------------------------
// WMMA 16x16x32 bf16 per-lane fragment layout (CDNA5 ISA 7.12.2):
//   lane L holds row/col (L & 15); kbase = (L>>4)*8
//   elements 0..7  <-> K = kbase + 0..7        (contiguous 16B)
//   elements 8..15 <-> K = kbase + 16 + 0..7   (contiguous 16B)
// A (rows of X/H) and B (rows of W; we contract W along dim 1) share this
// pattern, so fragments load directly from global memory, no transpose.
// ---------------------------------------------------------------------------

__device__ __forceinline__ v16bf load_frag_bf16(const __bf16* __restrict__ p) {
    v8bf lo = *(const v8bf*)(p);
    v8bf hi = *(const v8bf*)(p + 16);
    return __builtin_shufflevector(lo, hi, 0, 1, 2, 3, 4, 5, 6, 7,
                                           8, 9, 10, 11, 12, 13, 14, 15);
}

__device__ __forceinline__ v16bf load_frag_f32(const float* __restrict__ p) {
    v16bf r;
#pragma unroll
    for (int e = 0; e < 8; ++e) r[e] = (__bf16)p[e];
#pragma unroll
    for (int e = 0; e < 8; ++e) r[8 + e] = (__bf16)p[16 + e];
    return r;
}

__device__ __forceinline__ v8f wmma_bf16(v16bf a, v16bf b, v8f c) {
    return __builtin_amdgcn_wmma_f32_16x16x32_bf16(
        /*neg_a=*/false, a, /*neg_b=*/false, b,
        /*c_mod=*/(short)0, c, /*reuse_a=*/false, /*reuse_b=*/false);
}

// ---------------------------------------------------------------------------
// XP precompute: XP[t][b][h] = x[b][t][:] . W_ih[h][:] + (b_ih[h]+b_hh[h])
// Block tile 64x64, 4 waves 2x2, each wave 32x32 (4 WMMA per K-chunk).
// grid = (HH/64, SS*(BB/64)) = (16, 2048)
// ---------------------------------------------------------------------------
__global__ __launch_bounds__(128)
void xp_gemm_kernel(const float*  __restrict__ x,     // [BB, SS, DD] fp32
                    const __bf16* __restrict__ Wih,   // [HH, DD] bf16
                    const float*  __restrict__ b_ih,  // [HH]
                    const float*  __restrict__ b_hh,  // [HH]
                    float*        __restrict__ xp)    // [SS, BB, HH] fp32
{
    const int lane  = threadIdx.x & 31;
    const int wave  = threadIdx.x >> 5;   // 0..3
    const int wm    = wave & 1;
    const int wn    = wave >> 1;
    const int lrow  = lane & 15;
    const int kbase = (lane >> 4) * 8;

    const int t  = blockIdx.y >> 2;
    const int mb = (blockIdx.y & 3) * 64;
    const int m0 = mb + wm * 32;
    const int n0 = blockIdx.x * 64 + wn * 32;

    v8f acc00 = {0.f,0.f,0.f,0.f,0.f,0.f,0.f,0.f};
    v8f acc01 = acc00, acc10 = acc00, acc11 = acc00;

    const float*  xr0 = x + (size_t)(m0 + lrow)      * (SS * DD) + (size_t)t * DD + kbase;
    const float*  xr1 = x + (size_t)(m0 + 16 + lrow) * (SS * DD) + (size_t)t * DD + kbase;
    const __bf16* wr0 = Wih + (size_t)(n0 + lrow)      * DD + kbase;
    const __bf16* wr1 = Wih + (size_t)(n0 + 16 + lrow) * DD + kbase;

#pragma unroll 2
    for (int k0 = 0; k0 < DD; k0 += 32) {
        v16bf a0 = load_frag_f32(xr0 + k0);
        v16bf a1 = load_frag_f32(xr1 + k0);
        v16bf b0 = load_frag_bf16(wr0 + k0);
        v16bf b1 = load_frag_bf16(wr1 + k0);
        acc00 = wmma_bf16(a0, b0, acc00);
        acc01 = wmma_bf16(a0, b1, acc01);
        acc10 = wmma_bf16(a1, b0, acc10);
        acc11 = wmma_bf16(a1, b1, acc11);
    }

    // D layout: lane L -> col (L&15); VGPR r -> row r + (L>>4)*8
#pragma unroll
    for (int j = 0; j < 2; ++j) {
        const int   ncol = n0 + j * 16 + lrow;
        const float bj   = b_ih[ncol] + b_hh[ncol];
        const v8f&  aj0  = j ? acc01 : acc00;
        const v8f&  aj1  = j ? acc11 : acc10;
#pragma unroll
        for (int r = 0; r < 8; ++r) {
            const int row0 = m0 + kbase + r;
            xp[((size_t)t * BB + row0)      * HH + ncol] = aj0[r] + bj;
            xp[((size_t)t * BB + row0 + 16) * HH + ncol] = aj1[r] + bj;
        }
    }
}

// ---------------------------------------------------------------------------
// Persistent scan: ONE launch runs all SS steps.
//   h_{t+1} = tanh(XP_t + h_t @ W_hh^T)
// 32 blocks (8 N-tiles x 4 M-tiles) of 256 threads (8 waves, 2x4, each wave
// a 32x32 tile). Steps are separated by a device-scope software grid barrier
// (monotone atomic counter; release on signal, acquire while spinning).
// h ping-pongs between two global buffers, so there are no write/read
// conflicts within a step and one barrier per step suffices.
// ---------------------------------------------------------------------------
__global__ __launch_bounds__(256)
void rnn_scan_persistent_kernel(const float*  __restrict__ XP,   // [SS, BB, HH] fp32
                                const __bf16* __restrict__ Whh,  // [HH, HH] bf16
                                __bf16* __restrict__ hA,         // [BB, HH] (h0, zeroed)
                                __bf16* __restrict__ hB,         // [BB, HH]
                                unsigned* __restrict__ bar)      // init 0
{
    const int lane  = threadIdx.x & 31;
    const int wave  = threadIdx.x >> 5;   // 0..7
    const int wm    = wave & 1;           // 2 waves in M
    const int wn    = wave >> 1;          // 4 waves in N
    const int lrow  = lane & 15;
    const int kbase = (lane >> 4) * 8;

    const int m0 = blockIdx.y * 64 + wm * 32;    // gridDim.y = 4
    const int n0 = blockIdx.x * 128 + wn * 32;   // gridDim.x = 8

    // Weight fragment base pointers are loop-invariant: this block re-reads
    // the same 128 W_hh rows every step (WGP$/L2 resident).
    const __bf16* wr0 = Whh + (size_t)(n0 + lrow)      * HH + kbase;
    const __bf16* wr1 = Whh + (size_t)(n0 + 16 + lrow) * HH + kbase;

    const __bf16* hin  = hA;
    __bf16*       hout = hB;

    for (int t = 0; t < SS; ++t) {
        v8f acc00 = {0.f,0.f,0.f,0.f,0.f,0.f,0.f,0.f};
        v8f acc01 = acc00, acc10 = acc00, acc11 = acc00;

        const __bf16* hr0 = hin + (size_t)(m0 + lrow)      * HH + kbase;
        const __bf16* hr1 = hin + (size_t)(m0 + 16 + lrow) * HH + kbase;

#pragma unroll 4
        for (int k0 = 0; k0 < HH; k0 += 32) {
            v16bf a0 = load_frag_bf16(hr0 + k0);
            v16bf a1 = load_frag_bf16(hr1 + k0);
            v16bf b0 = load_frag_bf16(wr0 + k0);
            v16bf b1 = load_frag_bf16(wr1 + k0);
            acc00 = wmma_bf16(a0, b0, acc00);
            acc01 = wmma_bf16(a0, b1, acc01);
            acc10 = wmma_bf16(a1, b0, acc10);
            acc11 = wmma_bf16(a1, b1, acc11);
        }

        const float* xpt = XP + (size_t)t * BB * HH;
#pragma unroll
        for (int j = 0; j < 2; ++j) {
            const int  ncol = n0 + j * 16 + lrow;
            const v8f& aj0  = j ? acc01 : acc00;
            const v8f& aj1  = j ? acc11 : acc10;
#pragma unroll
            for (int r = 0; r < 8; ++r) {
                const int row0 = m0 + kbase + r;
                const int row1 = row0 + 16;
                float v0 = tanhf(aj0[r] + xpt[(size_t)row0 * HH + ncol]);
                float v1 = tanhf(aj1[r] + xpt[(size_t)row1 * HH + ncol]);
                hout[(size_t)row0 * HH + ncol] = (__bf16)v0;
                hout[(size_t)row1 * HH + ncol] = (__bf16)v1;
            }
        }

        // ---- device-scope grid barrier between steps ----
        __threadfence();          // each thread drains its h stores to agent scope
        __syncthreads();          // whole block's stores are now agent-visible
        if (threadIdx.x == 0) {
            __hip_atomic_fetch_add(bar, 1u, __ATOMIC_RELEASE,
                                   __HIP_MEMORY_SCOPE_AGENT);
            const unsigned target = (unsigned)(t + 1) * NBLK;
            while (__hip_atomic_load(bar, __ATOMIC_ACQUIRE,
                                     __HIP_MEMORY_SCOPE_AGENT) < target)
                __builtin_amdgcn_s_sleep(1);
        }
        __syncthreads();

        const __bf16* tmp = hin; hin = hout; hout = (__bf16*)tmp;
    }
    // SS = 512 (even) -> final hidden state is in hA.
}

// ---------------------------------------------------------------------------
// Scan step (fused fallback, only if ws too small for XP):
// hout = tanh(x_t @ W_ih^T + hin @ W_hh^T + bias), one launch per step.
// ---------------------------------------------------------------------------
__global__ __launch_bounds__(128)
void rnn_step_fused_kernel(const float*  __restrict__ xt,    // x + t*DD ; row stride SS*DD
                           const __bf16* __restrict__ Wih,   // [HH, DD]
                           const __bf16* __restrict__ Whh,   // [HH, HH]
                           const float*  __restrict__ bias,  // [HH]
                           const __bf16* __restrict__ hin,   // [BB, HH]
                           __bf16*       __restrict__ hout)  // [BB, HH]
{
    const int lane  = threadIdx.x & 31;
    const int wave  = threadIdx.x >> 5;
    const int lrow  = lane & 15;
    const int kbase = (lane >> 4) * 8;

    const int m0 = blockIdx.y * 32;
    const int n0 = blockIdx.x * 64 + wave * 16;

    v8f acc0 = {0.f,0.f,0.f,0.f,0.f,0.f,0.f,0.f};
    v8f acc1 = acc0;

    const float*  xr0 = xt + (size_t)(m0 + lrow)      * (SS * DD) + kbase;
    const float*  xr1 = xt + (size_t)(m0 + 16 + lrow) * (SS * DD) + kbase;
    const __bf16* wr  = Wih + (size_t)(n0 + lrow) * DD + kbase;
#pragma unroll 2
    for (int k0 = 0; k0 < DD; k0 += 32) {
        v16bf a0 = load_frag_f32(xr0 + k0);
        v16bf a1 = load_frag_f32(xr1 + k0);
        v16bf b  = load_frag_bf16(wr + k0);
        acc0 = wmma_bf16(a0, b, acc0);
        acc1 = wmma_bf16(a1, b, acc1);
    }

    const __bf16* hr0 = hin + (size_t)(m0 + lrow)      * HH + kbase;
    const __bf16* hr1 = hin + (size_t)(m0 + 16 + lrow) * HH + kbase;
    const __bf16* wr2 = Whh + (size_t)(n0 + lrow) * HH + kbase;
#pragma unroll 2
    for (int k0 = 0; k0 < HH; k0 += 32) {
        v16bf a0 = load_frag_bf16(hr0 + k0);
        v16bf a1 = load_frag_bf16(hr1 + k0);
        v16bf b  = load_frag_bf16(wr2 + k0);
        acc0 = wmma_bf16(a0, b, acc0);
        acc1 = wmma_bf16(a1, b, acc1);
    }

    const int   ncol = n0 + lrow;
    const float bj   = bias[ncol];
    const int   mr   = m0 + kbase;
#pragma unroll
    for (int r = 0; r < 8; ++r) {
        float v0 = tanhf(acc0[r] + bj);
        float v1 = tanhf(acc1[r] + bj);
        hout[(size_t)(mr + r)      * HH + ncol] = (__bf16)v0;
        hout[(size_t)(mr + r + 16) * HH + ncol] = (__bf16)v1;
    }
}

// ---------------------------------------------------------------------------
// Head: out = h @ W_head^T + b_head  -> fp32 [BB, DD]
// grid = (DD/64, BB/64) = (8, 4), 128 threads, wave tile 32x32.
// ---------------------------------------------------------------------------
__global__ __launch_bounds__(128)
void head_kernel(const __bf16* __restrict__ h,     // [BB, HH] bf16
                 const __bf16* __restrict__ Whd,   // [DD, HH] bf16
                 const float*  __restrict__ bhd,   // [DD]
                 float*        __restrict__ out)   // [BB, DD] fp32
{
    const int lane  = threadIdx.x & 31;
    const int wave  = threadIdx.x >> 5;
    const int wm    = wave & 1;
    const int wn    = wave >> 1;
    const int lrow  = lane & 15;
    const int kbase = (lane >> 4) * 8;

    const int m0 = blockIdx.y * 64 + wm * 32;
    const int n0 = blockIdx.x * 64 + wn * 32;

    v8f acc00 = {0.f,0.f,0.f,0.f,0.f,0.f,0.f,0.f};
    v8f acc01 = acc00, acc10 = acc00, acc11 = acc00;

    const __bf16* hr0 = h + (size_t)(m0 + lrow)      * HH + kbase;
    const __bf16* hr1 = h + (size_t)(m0 + 16 + lrow) * HH + kbase;
    const __bf16* wr0 = Whd + (size_t)(n0 + lrow)      * HH + kbase;
    const __bf16* wr1 = Whd + (size_t)(n0 + 16 + lrow) * HH + kbase;

#pragma unroll 4
    for (int k0 = 0; k0 < HH; k0 += 32) {
        v16bf a0 = load_frag_bf16(hr0 + k0);
        v16bf a1 = load_frag_bf16(hr1 + k0);
        v16bf b0 = load_frag_bf16(wr0 + k0);
        v16bf b1 = load_frag_bf16(wr1 + k0);
        acc00 = wmma_bf16(a0, b0, acc00);
        acc01 = wmma_bf16(a0, b1, acc01);
        acc10 = wmma_bf16(a1, b0, acc10);
        acc11 = wmma_bf16(a1, b1, acc11);
    }

#pragma unroll
    for (int j = 0; j < 2; ++j) {
        const int   ncol = n0 + j * 16 + lrow;
        const float bj   = bhd[ncol];
        const v8f&  aj0  = j ? acc01 : acc00;
        const v8f&  aj1  = j ? acc11 : acc10;
#pragma unroll
        for (int r = 0; r < 8; ++r) {
            const int row0 = m0 + kbase + r;
            out[(size_t)row0 * DD + ncol]        = aj0[r] + bj;
            out[(size_t)(row0 + 16) * DD + ncol] = aj1[r] + bj;
        }
    }
}

// ---------------------------------------------------------------------------
// Prep: fp32 -> bf16 weight conversion; bias fuse; h0 = 0; barrier = 0
// ---------------------------------------------------------------------------
__global__ void cvt_f32_bf16_kernel(const float* __restrict__ src,
                                    __bf16* __restrict__ dst, int n) {
    int i = blockIdx.x * blockDim.x + threadIdx.x;
    if (i < n) dst[i] = (__bf16)src[i];
}

__global__ void prep_bias_h0_kernel(const float* __restrict__ b_ih,
                                    const float* __restrict__ b_hh,
                                    float* __restrict__ bias,
                                    __bf16* __restrict__ h0,
                                    unsigned* __restrict__ bar) {
    int i = blockIdx.x * blockDim.x + threadIdx.x;
    if (i == 0) *bar = 0u;                 // re-armed every call (replay-safe)
    if (i < HH) bias[i] = b_ih[i] + b_hh[i];
    for (int j = i; j < BB * HH; j += gridDim.x * blockDim.x)
        h0[j] = (__bf16)0.0f;
}

// ---------------------------------------------------------------------------
extern "C" void kernel_launch(void* const* d_in, const int* in_sizes, int n_in,
                              void* d_out, int out_size, void* d_ws, size_t ws_size,
                              hipStream_t stream) {
    const float* x      = (const float*)d_in[0];
    const float* W_ih   = (const float*)d_in[1];
    const float* W_hh   = (const float*)d_in[2];
    const float* b_ih   = (const float*)d_in[3];
    const float* b_hh   = (const float*)d_in[4];
    const float* W_head = (const float*)d_in[5];
    const float* b_head = (const float*)d_in[6];
    float* out = (float*)d_out;

    // Workspace carve-out. Fixed region ~5 MB; XP adds 512 MB fp32.
    char* ws = (char*)d_ws;
    __bf16*   Wih_bf = (__bf16*)ws;   ws += (size_t)HH * DD * 2;  // 1 MB
    __bf16*   Whh_bf = (__bf16*)ws;   ws += (size_t)HH * HH * 2;  // 2 MB
    __bf16*   Whd_bf = (__bf16*)ws;   ws += (size_t)DD * HH * 2;  // 1 MB
    float*    bias   = (float*)  ws;  ws += (size_t)HH * 4;
    unsigned* bar    = (unsigned*)ws; ws += 256;                  // barrier counter
    __bf16*   h0     = (__bf16*)ws;   ws += (size_t)BB * HH * 2;  // 512 KB
    __bf16*   h1     = (__bf16*)ws;   ws += (size_t)BB * HH * 2;  // 512 KB
    float*    XP     = (float*) ws;

    const size_t fixed_bytes = (size_t)(ws - (char*)d_ws);
    const size_t xp_bytes    = (size_t)SS * BB * HH * 4;          // 512 MB
    const bool   use_xp      = ws_size >= fixed_bytes + xp_bytes;

    // Prep: bf16 weights live in L2 for the whole scan (3 MB << 192 MB)
    cvt_f32_bf16_kernel<<<(HH * DD + 255) / 256, 256, 0, stream>>>(W_ih, Wih_bf, HH * DD);
    cvt_f32_bf16_kernel<<<(HH * HH + 255) / 256, 256, 0, stream>>>(W_hh, Whh_bf, HH * HH);
    cvt_f32_bf16_kernel<<<(DD * HH + 255) / 256, 256, 0, stream>>>(W_head, Whd_bf, DD * HH);
    prep_bias_h0_kernel<<<1024, 256, 0, stream>>>(b_ih, b_hh, bias, h0, bar);

    if (use_xp) {
        // One big parallel WMMA GEMM hoists the input projection (+bias) out
        // of the sequential scan: XP[t][b][h], fp32.
        dim3 gxp(HH / 64, SS * (BB / 64));  // (16, 2048)
        xp_gemm_kernel<<<gxp, 128, 0, stream>>>(x, Wih_bf, b_ih, b_hh, XP);

        // Single persistent launch for the whole 512-step scan.
        dim3 gscan(HH / 128, BB / 64);      // (8, 4) = NBLK = 32 blocks
        rnn_scan_persistent_kernel<<<gscan, 256, 0, stream>>>(XP, Whh_bf, h0, h1, bar);
    } else {
        // Fallback: fused per-step projection, one launch per step.
        dim3 gstep(HH / 64, BB / 32);       // (16, 8) = 128 blocks
        for (int t = 0; t < SS; ++t) {
            const __bf16* hin  = (t & 1) ? h1 : h0;
            __bf16*       hout = (t & 1) ? h0 : h1;
            rnn_step_fused_kernel<<<gstep, 128, 0, stream>>>(x + (size_t)t * DD,
                                                             Wih_bf, Whh_bf, bias,
                                                             hin, hout);
        }
    }
    // SS = 512 is even -> final hidden state lands in h0 (both paths).

    dim3 ghead(DD / 64, BB / 64);           // (8, 4)
    head_kernel<<<ghead, 128, 0, stream>>>(h0, Whd_bf, b_head, out);
}
